// BiLSTM_CRF_38070590111946
// MI455X (gfx1250) — matrix operand order repaired and checked by reading, not compiled, not run on hardware
//
#include <hip/hip_runtime.h>
#include <hip/hip_bf16.h>
#include <cstddef>

// ---------------------------------------------------------------------------
// BiLSTM-CRF forward for MI455X (gfx1250, wave32, WMMA bf16 f32-accum)
// ---------------------------------------------------------------------------

typedef __attribute__((ext_vector_type(16))) __bf16 v16bf;
typedef __attribute__((ext_vector_type(8)))  float  v8f;

union FragBF { v16bf v; unsigned u[8]; };

__device__ __forceinline__ unsigned short f2bf(float x) {
  unsigned u = __float_as_uint(x);
  unsigned r = (u + 0x7FFFu + ((u >> 16) & 1u)) >> 16;
  return (unsigned short)r;
}
__device__ __forceinline__ float sigm(float x) { return 1.0f / (1.0f + __expf(-x)); }

// ---------------------------------------------------------------------------
// K1: f32 -> bf16 convert (weights)
// ---------------------------------------------------------------------------
__global__ void f2bf_kernel(const float* __restrict__ in, unsigned short* __restrict__ out, int n) {
  int i = blockIdx.x * 256 + threadIdx.x;
  if (i < n) out[i] = f2bf(in[i]);
}

// ---------------------------------------------------------------------------
// K2: embedding gather to bf16.  Storage row q (== flat index of both the
// [B,L] gather and the [L,B] reshape view) -> emb[sentences_flat[q]].
// ---------------------------------------------------------------------------
__global__ void embed_gather(const int* __restrict__ sent, const float* __restrict__ emb,
                             unsigned short* __restrict__ out) {
  int r = blockIdx.x;          // 0..65535
  int f = threadIdx.x;         // 0..255
  int tok = sent[r];
  out[(size_t)r * 256 + f] = f2bf(emb[(size_t)tok * 256 + f]);
}

// ---------------------------------------------------------------------------
// K3: Out[M,N](f32) = A[M,K](bf16) * W[N,K]^T(bf16) + b1[N] + b2[N]
// One wave computes a 16(M) x 64(N) strip; A fragment reused over 4 B tiles.
// Fragment loads vectorize to global_load_b128 pairs (verified in asm).
// ---------------------------------------------------------------------------
__global__ __launch_bounds__(256) void gemm_bf16_bias(
    const unsigned short* __restrict__ A, const unsigned short* __restrict__ W,
    const float* __restrict__ b1, const float* __restrict__ b2,
    float* __restrict__ Out, int M, int N, int K) {
  const int lane = threadIdx.x & 31;
  const int wid  = (blockIdx.x * 256 + threadIdx.x) >> 5;
  const int stripsN = N >> 6;
  const int tm = wid / stripsN;
  const int n0 = (wid % stripsN) << 6;
  if (tm * 16 >= M) return;
  const int half = lane >> 4, nl = lane & 15;

  v8f acc[4];
#pragma unroll
  for (int q = 0; q < 4; ++q)
#pragma unroll
    for (int r = 0; r < 8; ++r) acc[q][r] = 0.0f;

  for (int k0 = 0; k0 < K; k0 += 32) {
    FragBF fa;
    const unsigned short* arow = A + (size_t)(tm * 16 + nl) * K + k0;
#pragma unroll
    for (int v = 0; v < 8; ++v) {
      int kb = ((v >= 4) ? 16 : 0) + half * 8 + 2 * (v & 3);
      fa.u[v] = *(const unsigned*)(arow + kb);
    }
#pragma unroll
    for (int q = 0; q < 4; ++q) {
      FragBF fb;
      const unsigned short* brow = W + (size_t)(n0 + q * 16 + nl) * K + k0;
#pragma unroll
      for (int v = 0; v < 8; ++v) {
        int kb = ((v >= 4) ? 16 : 0) + half * 8 + 2 * (v & 3);
        fb.u[v] = *(const unsigned*)(brow + kb);
      }
      acc[q] = __builtin_amdgcn_wmma_f32_16x16x32_bf16(false, fa.v, false, fb.v,
                                                       (short)0, acc[q], false, false);
    }
  }
#pragma unroll
  for (int q = 0; q < 4; ++q) {
    const int col = n0 + q * 16 + nl;
    const float bias = (b1 ? b1[col] : 0.0f) + (b2 ? b2[col] : 0.0f);
#pragma unroll
    for (int r = 0; r < 8; ++r)
      Out[(size_t)(tm * 16 + r + half * 8) * N + col] = acc[q][r] + bias;
  }
}

// ---------------------------------------------------------------------------
// K4: persistent recurrent LSTM.  gridDim.x = 2 (dir 0 = fwd, 1 = bwd).
// X*   : precomputed x@Wih^T + bih + bhh, f32 [L,64,512] (row l*64+b)
// Whh* : bf16 [512,128]
// h0,c0: f32 [2,64,128];  Y: f32 [L,64,256] (dir writes cols dir*128..)
// cfinal: f32 [2,64,128] or nullptr
//
// 1024 threads = 32 waves; each wave owns ONE (m_tile,u_tile) group with all
// four gate tiles (i/f/g/o), so the nonlinearity needs no cross-lane traffic.
// The 16 Whh B-fragments a wave needs are loaded ONCE and pinned in VGPRs for
// all 1024 timesteps: the steady-state step is X loads + 8 ds_load_b128 (h
// fragments from LDS) + 16 v_wmma + gate VALU + 2 barriers.  Next step's X
// slab (128 KB) is prefetched via global_prefetch so the accumulator-init
// loads hit WGP$/L2.
// ---------------------------------------------------------------------------
#define HS 132  // padded LDS row stride (bf16 elems) to avoid bank conflicts

__global__ __launch_bounds__(1024, 1) void lstm_rec(
    const float* __restrict__ Xf, const float* __restrict__ Xb,
    const unsigned short* __restrict__ Whhf, const unsigned short* __restrict__ Whhb,
    const float* __restrict__ h0, const float* __restrict__ c0,
    float* __restrict__ Y, float* __restrict__ cfinal) {
  const int dir = blockIdx.x;
  const float* X = dir ? Xb : Xf;
  const unsigned short* Whh = dir ? Whhb : Whhf;

  __shared__ unsigned short hL[64 * HS];

  const int tid = threadIdx.x, lane = tid & 31, w = tid >> 5;  // 32 waves
  const int half = lane >> 4, nl = lane & 15;

  for (int i = tid; i < 64 * 128; i += 1024) {
    int b = i >> 7, u = i & 127;
    hL[b * HS + u] = f2bf(h0[(dir * 64 + b) * 128 + u]);
  }

  const int mt = w >> 3;   // 0..3  batch tile
  const int ut = w & 7;    // 0..7  hidden-unit tile
  const int mrow = mt * 16 + half * 8;
  const int ucol = ut * 16 + nl;

  v8f ct;
#pragma unroll
  for (int r = 0; r < 8; ++r)
    ct[r] = c0[(dir * 64 + mrow + r) * 128 + ucol];

  // Pin the 16 Whh B-fragments (4 gates x 4 k-steps) in registers.
  v16bf wc[4][4];
#pragma unroll
  for (int q = 0; q < 4; ++q)
#pragma unroll
    for (int k = 0; k < 4; ++k) {
      FragBF fb;
      const unsigned short* brow = Whh + (size_t)(q * 128 + ucol) * 128 + k * 32;
#pragma unroll
      for (int v = 0; v < 8; ++v)
        fb.u[v] = *(const unsigned*)(brow + ((v >= 4) ? 16 : 0) + half * 8 + 2 * (v & 3));
      wc[q][k] = fb.v;
    }

  __syncthreads();

  for (int t = 0; t < 1024; ++t) {
    const int l = dir ? (1023 - t) : t;
    const float* Xl = X + (size_t)l * 64 * 512;

    // Prefetch next step's X slab (128 KB; 1024 threads x 128 B stride).
    if (t + 1 < 1024) {
      const int ln = dir ? (1022 - t) : (t + 1);
      __builtin_prefetch((const char*)(X + (size_t)ln * 64 * 512) + tid * 128, 0, 3);
    }

    v8f acc[4];
#pragma unroll
    for (int q = 0; q < 4; ++q) {
      const int col = q * 128 + ucol;
#pragma unroll
      for (int r = 0; r < 8; ++r)
        acc[q][r] = Xl[(size_t)(mrow + r) * 512 + col];
    }

#pragma unroll
    for (int k = 0; k < 4; ++k) {
      FragBF fa;
      const unsigned short* arow = &hL[(mt * 16 + nl) * HS + k * 32];
#pragma unroll
      for (int v = 0; v < 8; ++v) {
        int kb = ((v >= 4) ? 16 : 0) + half * 8 + 2 * (v & 3);
        fa.u[v] = *(const unsigned*)(arow + kb);
      }
#pragma unroll
      for (int q = 0; q < 4; ++q)
        acc[q] = __builtin_amdgcn_wmma_f32_16x16x32_bf16(false, fa.v, false, wc[q][k],
                                                         (short)0, acc[q], false, false);
    }

    float hnew[8];
#pragma unroll
    for (int r = 0; r < 8; ++r) {
      float ig = sigm(acc[0][r]);
      float fg = sigm(acc[1][r]);
      float gg = tanhf(acc[2][r]);
      float og = sigm(acc[3][r]);
      float cv = fg * ct[r] + ig * gg;
      ct[r] = cv;
      hnew[r] = og * tanhf(cv);
    }

    __syncthreads();  // all LDS reads of old h complete
#pragma unroll
    for (int r = 0; r < 8; ++r) {
      hL[(mrow + r) * HS + ucol] = f2bf(hnew[r]);
      Y[((size_t)l * 64 + mrow + r) * 256 + dir * 128 + ucol] = hnew[r];
    }
    __syncthreads();  // new h visible to all waves
  }

  if (cfinal) {
#pragma unroll
    for (int r = 0; r < 8; ++r)
      cfinal[(dir * 64 + mrow + r) * 128 + ucol] = ct[r];
  }
}

// ---------------------------------------------------------------------------
// K5: attention pieces
// ---------------------------------------------------------------------------
__global__ void catc_build(const float* __restrict__ cfb, float* __restrict__ catc) {
  int i = blockIdx.x * 256 + threadIdx.x;  // 64*256
  int b = i >> 8, f = i & 255;
  catc[i] = (f < 128) ? cfb[(0 * 64 + b) * 128 + f] : cfb[(64 + b) * 128 + (f - 128)];
}

__global__ void attn_e(const float* __restrict__ out1, const float* __restrict__ catc,
                       float* __restrict__ e) {
  int b = blockIdx.x;
  int l = blockIdx.y * 256 + threadIdx.x;
  __shared__ float cc[256];
  cc[threadIdx.x] = catc[b * 256 + threadIdx.x];
  __syncthreads();
  const float* row = out1 + ((size_t)l * 64 + b) * 256;
  float s = 0.0f;
#pragma unroll 8
  for (int f = 0; f < 256; ++f) s += row[f] * cc[f];
  e[b * 1024 + l] = s;
}

__global__ void attn_softmax(const float* __restrict__ e, float* __restrict__ s) {
  int b = blockIdx.x, tid = threadIdx.x;
  __shared__ float red[256];
  const float* eb = e + b * 1024;
  float m = -3.4e38f;
  for (int l = tid; l < 1024; l += 256) m = fmaxf(m, eb[l]);
  red[tid] = m; __syncthreads();
  for (int st = 128; st > 0; st >>= 1) {
    if (tid < st) red[tid] = fmaxf(red[tid], red[tid + st]);
    __syncthreads();
  }
  float mx = red[0]; __syncthreads();
  float sum = 0.0f;
  for (int l = tid; l < 1024; l += 256) sum += expf(eb[l] - mx);
  red[tid] = sum; __syncthreads();
  for (int st = 128; st > 0; st >>= 1) {
    if (tid < st) red[tid] += red[tid + st];
    __syncthreads();
  }
  float inv = 1.0f / red[0];
  for (int l = tid; l < 1024; l += 256) s[b * 1024 + l] = expf(eb[l] - mx) * inv;
}

// gi[l,b, 0:256]   = out1[l,b,:]               (bf16)
// gi[l,b, 256:512] = 64 * s[b,l] * catc[b,:]   (bf16)
__global__ void gi_build(const float* __restrict__ out1, const float* __restrict__ s,
                         const float* __restrict__ catc, unsigned short* __restrict__ gi) {
  int r = blockIdx.x;      // 0..65535 (== l*64+b)
  int f = threadIdx.x;     // 0..255
  int b = r & 63, l = r >> 6;
  float o = out1[(size_t)r * 256 + f];
  float a = 64.0f * s[b * 1024 + l] * catc[b * 256 + f];
  gi[(size_t)r * 512 + f]       = f2bf(o);
  gi[(size_t)r * 512 + 256 + f] = f2bf(a);
}

// ---------------------------------------------------------------------------
// K6: feats[r,c] = out2[r,:] . Wout[c,:] + bout[c]   (C = 5)
// ---------------------------------------------------------------------------
__global__ void proj_out(const float* __restrict__ out2, const float* __restrict__ Wout,
                         const float* __restrict__ bout, float* __restrict__ feats) {
  __shared__ float wsm[5 * 256];
  int tid = threadIdx.x;
  for (int i = tid; i < 5 * 256; i += 256) wsm[i] = Wout[i];
  __syncthreads();
  int r = blockIdx.x * 256 + tid;  // 0..65535
  const float* row = out2 + (size_t)r * 256;
#pragma unroll
  for (int c = 0; c < 5; ++c) {
    float a = bout[c];
    const float* wc = &wsm[c * 256];
#pragma unroll 8
    for (int f = 0; f < 256; ++f) a += row[f] * wc[f];
    feats[(size_t)r * 5 + c] = a;
  }
}

// ---------------------------------------------------------------------------
// K7: CRF forward algorithm.  One thread per batch element.
// feats row index = b*1024 + l (storage-preserving reshape).
// ---------------------------------------------------------------------------
__global__ void crf_fwd(const float* __restrict__ feats, const float* __restrict__ masks,
                        const float* __restrict__ trans, float* __restrict__ out) {
  int b = threadIdx.x;
  if (b >= 64) return;
  float tr[25];
#pragma unroll
  for (int i = 0; i < 25; ++i) tr[i] = trans[i];
  float sc[5];
#pragma unroll
  for (int i = 0; i < 5; ++i) sc[i] = -10000.0f;
  sc[3] = 0.0f;  // START
  for (int l = 0; l < 1024; ++l) {
    const float* ft = feats + ((size_t)b * 1024 + l) * 5;
    float mt = masks[b * 1024 + l];
    float st[5];
#pragma unroll
    for (int i = 0; i < 5; ++i) {
      float m = -3.4e38f;
#pragma unroll
      for (int j = 0; j < 5; ++j) m = fmaxf(m, sc[j] + tr[i * 5 + j]);
      float ssum = 0.0f;
#pragma unroll
      for (int j = 0; j < 5; ++j) ssum += expf(sc[j] + tr[i * 5 + j] - m);
      st[i] = ft[i] + m + logf(ssum);
    }
#pragma unroll
    for (int i = 0; i < 5; ++i) sc[i] = mt * st[i] + (1.0f - mt) * sc[i];
  }
  float m = -3.4e38f;
#pragma unroll
  for (int i = 0; i < 5; ++i) m = fmaxf(m, sc[i] + tr[4 * 5 + i]);  // STOP = 4
  float ssum = 0.0f;
#pragma unroll
  for (int i = 0; i < 5; ++i) ssum += expf(sc[i] + tr[4 * 5 + i] - m);
  out[b] = m + logf(ssum);
}

// ---------------------------------------------------------------------------
// Host orchestration
// ---------------------------------------------------------------------------
static void launch_gemm(const unsigned short* A, const unsigned short* W,
                        const float* b1, const float* b2, float* Out,
                        int M, int N, int K, hipStream_t s) {
  int waves = (M / 16) * (N / 64);
  int blocks = (waves + 7) / 8;
  gemm_bf16_bias<<<blocks, 256, 0, s>>>(A, W, b1, b2, Out, M, N, K);
}

extern "C" void kernel_launch(void* const* d_in, const int* in_sizes, int n_in,
                              void* d_out, int out_size, void* d_ws, size_t ws_size,
                              hipStream_t stream) {
  const int*   sent  = (const int*)d_in[0];
  const float* masks = (const float*)d_in[1];
  const float* emb   = (const float*)d_in[2];
  const float* Wih1f = (const float*)d_in[3];
  const float* Whh1f = (const float*)d_in[4];
  const float* bih1f = (const float*)d_in[5];
  const float* bhh1f = (const float*)d_in[6];
  const float* Wih1b = (const float*)d_in[7];
  const float* Whh1b = (const float*)d_in[8];
  const float* bih1b = (const float*)d_in[9];
  const float* bhh1b = (const float*)d_in[10];
  const float* Wih2f = (const float*)d_in[11];
  const float* Whh2f = (const float*)d_in[12];
  const float* bih2f = (const float*)d_in[13];
  const float* bhh2f = (const float*)d_in[14];
  const float* Wih2b = (const float*)d_in[15];
  const float* Whh2b = (const float*)d_in[16];
  const float* bih2b = (const float*)d_in[17];
  const float* bhh2b = (const float*)d_in[18];
  const float* Wout  = (const float*)d_in[19];
  const float* bout  = (const float*)d_in[20];
  const float* trans = (const float*)d_in[21];
  const float* h0    = (const float*)d_in[22];
  const float* c0    = (const float*)d_in[23];
  const float* h02   = (const float*)d_in[24];
  const float* c02   = (const float*)d_in[25];
  float* out = (float*)d_out;

  char* ws = (char*)d_ws;
  size_t off = 0;
  auto take = [&](size_t bytes) -> char* {
    char* p = ws + off;
    off += (bytes + 255) & ~(size_t)255;
    return p;
  };

  const size_t R = 65536;  // L*B rows
  float* Xf    = (float*)take(R * 512 * 4);          // 128 MB (reused by layer 2)
  float* Xb    = (float*)take(R * 512 * 4);          // 128 MB (reused by layer 2)
  float* out12 = (float*)take(R * 256 * 4);          // 64 MB (out1, reused as out2)
  unsigned short* giw  = (unsigned short*)take(R * 512 * 2);  // 64 MB (gi; reused as feats)
  unsigned short* embB = (unsigned short*)take(R * 256 * 2);  // 32 MB
  unsigned short* wIH1F = (unsigned short*)take(512 * 256 * 2);
  unsigned short* wIH1B = (unsigned short*)take(512 * 256 * 2);
  unsigned short* wHH1F = (unsigned short*)take(512 * 128 * 2);
  unsigned short* wHH1B = (unsigned short*)take(512 * 128 * 2);
  unsigned short* wIH2F = (unsigned short*)take(512 * 512 * 2);
  unsigned short* wIH2B = (unsigned short*)take(512 * 512 * 2);
  unsigned short* wHH2F = (unsigned short*)take(512 * 128 * 2);
  unsigned short* wHH2B = (unsigned short*)take(512 * 128 * 2);
  float* cfb   = (float*)take(2 * 64 * 128 * 4);
  float* catcB = (float*)take(64 * 256 * 4);
  float* ebuf  = (float*)take(64 * 1024 * 4);
  float* sbuf  = (float*)take(64 * 1024 * 4);
  float* feats = (float*)giw;  // reuse: gi dead once X2 GEMMs are done

  auto cvt = [&](const float* src, unsigned short* dst, int n) {
    f2bf_kernel<<<(n + 255) / 256, 256, 0, stream>>>(src, dst, n);
  };
  cvt(Wih1f, wIH1F, 512 * 256);
  cvt(Wih1b, wIH1B, 512 * 256);
  cvt(Whh1f, wHH1F, 512 * 128);
  cvt(Whh1b, wHH1B, 512 * 128);
  cvt(Wih2f, wIH2F, 512 * 512);
  cvt(Wih2b, wIH2B, 512 * 512);
  cvt(Whh2f, wHH2F, 512 * 128);
  cvt(Whh2b, wHH2B, 512 * 128);

  // embeddings (flat storage == reference's reshape view)
  embed_gather<<<65536, 256, 0, stream>>>(sent, emb, embB);

  // layer-1 input projections (bias folded in)
  launch_gemm(embB, wIH1F, bih1f, bhh1f, Xf, 65536, 512, 256, stream);
  launch_gemm(embB, wIH1B, bih1b, bhh1b, Xb, 65536, 512, 256, stream);

  // layer-1 recurrence (fwd + bwd concurrently), final cells -> cfb
  lstm_rec<<<2, 1024, 0, stream>>>(Xf, Xb, wHH1F, wHH1B, h0, c0, out12, cfb);

  // attention
  catc_build<<<64, 256, 0, stream>>>(cfb, catcB);
  attn_e<<<dim3(64, 4), 256, 0, stream>>>(out12, catcB, ebuf);
  attn_softmax<<<64, 256, 0, stream>>>(ebuf, sbuf);
  gi_build<<<65536, 256, 0, stream>>>(out12, sbuf, catcB, giw);

  // layer-2 input projections
  launch_gemm(giw, wIH2F, bih2f, bhh2f, Xf, 65536, 512, 512, stream);
  launch_gemm(giw, wIH2B, bih2b, bhh2b, Xb, 65536, 512, 512, stream);

  // layer-2 recurrence (out2 reuses out1 buffer)
  lstm_rec<<<2, 1024, 0, stream>>>(Xf, Xb, wHH2F, wHH2B, h02, c02, out12, nullptr);

  // output projection + CRF forward
  proj_out<<<256, 256, 0, stream>>>(out12, Wout, bout, feats);
  crf_fwd<<<1, 64, 0, stream>>>(feats, masks, trans, out);
}
